// WindowAttention_49589692399803
// MI455X (gfx1250) — matrix operand order repaired and checked by reading, not compiled, not run on hardware
//
#include <hip/hip_runtime.h>
#include <hip/hip_bf16.h>

// ---------------------------------------------------------------------------
// WindowAttention on MI455X (gfx1250): bf16 WMMA everywhere, f32 accumulate.
//   B_=256, N=196, C=256, H=8, hd=32, nW=64. scale = 32^-0.5.
// Pipeline:
//   0) one-pass f32->bf16 convert of x, qkv_w, proj_w into workspace
//   1) qkv GEMM (bf16 operands, 32x64 per wave), scatter q(scaled)/k/v
//   2) fused attention per (b,h,16-row tile): QK^T WMMA, bias+mask, softmax
//      in VGPRs (shfl_xor), PV WMMA via LDS
//   3) output projection GEMM (32x64 per wave), f32 epilogue to d_out
// ---------------------------------------------------------------------------

typedef __attribute__((ext_vector_type(16))) __bf16 v16bf;
typedef __attribute__((ext_vector_type(8)))  float  v8f;

union FragBF {
    v16bf  v;
    __bf16 h[16];
    uint4  q[2];
};

__device__ __forceinline__ v8f wmma_bf16(v16bf a, v16bf b, v8f c) {
    // D(f32 16x16) = A(16x32 bf16) * B(32x16 bf16) + C
    return __builtin_amdgcn_wmma_f32_16x16x32_bf16(false, a, false, b,
                                                   (short)0, c, false, false);
}

// A-operand fragment (16x32 bf16) from a bf16 row (row-major, K-contiguous).
// Lane l holds row M=l%16; half i -> K = (i&7) + 8*(2*(i>>3)+hi)  => 2 x b128
__device__ __forceinline__ v16bf loadA_bf16(const __bf16* __restrict__ rowp, int hi) {
    FragBF f;
    f.q[0] = ((const uint4*)(rowp + 8 * hi))[0];
    f.q[1] = ((const uint4*)(rowp + 16 + 8 * hi))[0];
    return f.v;
}

// B-operand fragment (32x16 bf16) from a bf16 row of W (W[n][k], K-contiguous).
// Lane l holds col N=l%16; half i -> K = (i&7) + 8*(2*(i>>3)+hi)  => 2 x b128
__device__ __forceinline__ v16bf loadB_bf16(const __bf16* __restrict__ rowp, int hi) {
    FragBF f;
    f.q[0] = ((const uint4*)(rowp + 8 * hi))[0];
    f.q[1] = ((const uint4*)(rowp + 16 + 8 * hi))[0];
    return f.v;
}

#define B_WIN   256
#define NTOK    196
#define CCH     256
#define NHEAD   8
#define HD      32
#define NROWS   50176          // B_ * N
#define SCALE_Q 0.17677669529663687f

// ---------------------------------------------------------------------------
// Kernel 0: one-pass f32 -> bf16 (8 elems/thread, b128 in / b128 out)
// ---------------------------------------------------------------------------
__global__ __launch_bounds__(256) void cvt_bf16_kernel(
    const float* __restrict__ src, __bf16* __restrict__ dst, int n8)
{
    const int i = blockIdx.x * 256 + threadIdx.x;
    if (i >= n8) return;
    const float4* s = (const float4*)(src) + 2 * (size_t)i;
    float4 a = s[0], b = s[1];
    FragBF f;   // use first 8 halves
    f.h[0] = (__bf16)a.x; f.h[1] = (__bf16)a.y; f.h[2] = (__bf16)a.z; f.h[3] = (__bf16)a.w;
    f.h[4] = (__bf16)b.x; f.h[5] = (__bf16)b.y; f.h[6] = (__bf16)b.z; f.h[7] = (__bf16)b.w;
    ((uint4*)dst)[i] = f.q[0];
}

// ---------------------------------------------------------------------------
// Kernel 1: qkv = x @ qkv_w.T + qkv_b  (50176 x 256) x (256 x 768), all bf16 in.
// One wave -> 32x64 tile (2 M-tiles share every B fragment).
// Scatter q(scaled)/k/v to bf16 [B,H,N,32] workspaces.
// ---------------------------------------------------------------------------
__global__ __launch_bounds__(32) void qkv_kernel(
    const __bf16* __restrict__ xb, const __bf16* __restrict__ wb,
    const float* __restrict__ qkv_b,
    __bf16* __restrict__ qws, __bf16* __restrict__ kws, __bf16* __restrict__ vws)
{
    const int lane = threadIdx.x & 31;
    const int hi = lane >> 4, lo = lane & 15;
    const int m0 = blockIdx.x * 32;
    const int n0 = blockIdx.y * 64;

    float acc[2][4][8];
#pragma unroll
    for (int mi = 0; mi < 2; ++mi)
#pragma unroll
        for (int j = 0; j < 4; ++j)
#pragma unroll
            for (int r = 0; r < 8; ++r) acc[mi][j][r] = 0.f;

#pragma unroll
    for (int kt = 0; kt < 8; ++kt) {                       // K = 256 = 8 * 32
        v16bf a0 = loadA_bf16(xb + (size_t)(m0 + lo) * CCH + kt * 32, hi);
        v16bf a1 = loadA_bf16(xb + (size_t)(m0 + 16 + lo) * CCH + kt * 32, hi);
#pragma unroll
        for (int j = 0; j < 4; ++j) {
            v16bf b = loadB_bf16(wb + (size_t)(n0 + j * 16 + lo) * CCH + kt * 32, hi);
            v8f c0, c1;
#pragma unroll
            for (int r = 0; r < 8; ++r) { c0[r] = acc[0][j][r]; c1[r] = acc[1][j][r]; }
            c0 = wmma_bf16(a0, b, c0);
            c1 = wmma_bf16(a1, b, c1);
#pragma unroll
            for (int r = 0; r < 8; ++r) { acc[0][j][r] = c0[r]; acc[1][j][r] = c1[r]; }
        }
    }

#pragma unroll
    for (int mi = 0; mi < 2; ++mi)
#pragma unroll
        for (int j = 0; j < 4; ++j) {
            const int c     = n0 + j * 16 + lo;            // 0..767
            const int which = c >> 8;                      // 0=q 1=k 2=v
            const int rem   = c & 255;
            const int hh    = rem >> 5, d = rem & 31;
            const float bias = qkv_b[c];
            const float scl  = (which == 0) ? SCALE_Q : 1.0f;
            __bf16* dst = (which == 0) ? qws : ((which == 1) ? kws : vws);
#pragma unroll
            for (int r = 0; r < 8; ++r) {
                const int tok = m0 + mi * 16 + r + 8 * hi; // always < 50176
                const int bb  = tok / NTOK;
                const int nn  = tok - bb * NTOK;
                const float v = (acc[mi][j][r] + bias) * scl;
                dst[(((size_t)bb * NHEAD + hh) * NTOK + nn) * HD + d] = (__bf16)v;
            }
        }
}

// ---------------------------------------------------------------------------
// Kernel 2: per (b, h, 16-row tile): QK^T (13 WMMA) + bias + mask, softmax in
// VGPRs (shfl_xor over column lanes), PV (14 WMMA) via LDS-staged P and V.
// ---------------------------------------------------------------------------
__global__ __launch_bounds__(32) void attn_kernel(
    const __bf16* __restrict__ qws, const __bf16* __restrict__ kws,
    const __bf16* __restrict__ vws, const float* __restrict__ mask,
    const float* __restrict__ bias_table, const int* __restrict__ rel_index,
    __bf16* __restrict__ ows)
{
    __shared__ __bf16 pbuf[16 * 224];      // probs, zero-padded K to 224
    __shared__ __bf16 vbuf[224 * 32];      // V rows, zero-padded to 224

    const int lane = threadIdx.x & 31;
    const int hi = lane >> 4, lo = lane & 15;
    const int mt = blockIdx.x;             // 0..12
    const int h  = blockIdx.y;             // 0..7
    const int b  = blockIdx.z;             // 0..255
    const int m0 = mt * 16;
    const size_t bh = (size_t)b * NHEAD + h;
    const __bf16* qb = qws + bh * NTOK * HD;
    const __bf16* kb = kws + bh * NTOK * HD;
    const __bf16* vb = vws + bh * NTOK * HD;

    // Stage V into LDS (each row = 64B = 4 x b128), zero-pad rows >= 196.
#pragma unroll
    for (int t = lane; t < 224 * 4; t += 32) {
        const int row = t >> 2, qi = t & 3;
        uint4 val = make_uint4(0u, 0u, 0u, 0u);
        if (row < NTOK) val = ((const uint4*)(vb + (size_t)row * HD))[qi];
        ((uint4*)(vbuf + row * HD))[qi] = val;
    }

    // Q fragment (A layout), straight from global bf16 with two b128 loads.
    int qrow = m0 + lo; if (qrow > NTOK - 1) qrow = NTOK - 1;
    FragBF qa;
    qa.q[0] = ((const uint4*)(qb + (size_t)qrow * HD + 8 * hi))[0];
    qa.q[1] = ((const uint4*)(qb + (size_t)qrow * HD + 16 + 8 * hi))[0];

    // --- QK^T: one WMMA per 16-col tile (hd=32 == one K step) ---
    float acc[13][8];
#pragma unroll
    for (int nt = 0; nt < 13; ++nt) {
        int krow = nt * 16 + lo; if (krow > NTOK - 1) krow = NTOK - 1;
        FragBF kf;                                   // B layout: lane=col, K contiguous
        kf.q[0] = ((const uint4*)(kb + (size_t)krow * HD + 8 * hi))[0];
        kf.q[1] = ((const uint4*)(kb + (size_t)krow * HD + 16 + 8 * hi))[0];
        v8f c = {0.f, 0.f, 0.f, 0.f, 0.f, 0.f, 0.f, 0.f};
        c = wmma_bf16(qa.v, kf.v, c);
#pragma unroll
        for (int r = 0; r < 8; ++r) {
            const int M  = m0 + r + 8 * hi;
            const int Nc = nt * 16 + lo;
            float s = -3.0e38f;                      // padded -> exp underflows to 0
            if (M < NTOK && Nc < NTOK) {
                const int idx = rel_index[M * NTOK + Nc];
                s = c[r] + bias_table[idx * NHEAD + h]
                         + mask[((size_t)(b & 63) * NTOK + M) * NTOK + Nc];
            }
            acc[nt][r] = s;
        }
    }

    // --- row softmax: reduce over nt (in-lane) and 16 column lanes (shfl_xor) ---
#pragma unroll
    for (int r = 0; r < 8; ++r) {
        float rm = -3.4e38f;
#pragma unroll
        for (int nt = 0; nt < 13; ++nt) rm = fmaxf(rm, acc[nt][r]);
#pragma unroll
        for (int s = 1; s < 16; s <<= 1) rm = fmaxf(rm, __shfl_xor(rm, s, 32));
        float sum = 0.f;
#pragma unroll
        for (int nt = 0; nt < 13; ++nt) {
            const float e = __expf(acc[nt][r] - rm);
            acc[nt][r] = e;
            sum += e;
        }
#pragma unroll
        for (int s = 1; s < 16; s <<= 1) sum += __shfl_xor(sum, s, 32);
        const float inv = 1.0f / sum;                // sum >= 1 (max elem contributes 1)
        const int prow = r + 8 * hi;
#pragma unroll
        for (int nt = 0; nt < 13; ++nt)
            pbuf[prow * 224 + nt * 16 + lo] = (__bf16)(acc[nt][r] * inv);
    }
    // zero-pad P columns 208..223
#pragma unroll
    for (int j2 = 0; j2 < 8; ++j2)
        pbuf[lo * 224 + 208 + hi * 8 + j2] = (__bf16)0.0f;

    __syncthreads();

    // --- PV: out(16x32) = P(16x224) x V(224x32), 7 K-steps x 2 col tiles ---
    float oacc[2][8];
#pragma unroll
    for (int od = 0; od < 2; ++od)
#pragma unroll
        for (int r = 0; r < 8; ++r) oacc[od][r] = 0.f;

#pragma unroll
    for (int kt = 0; kt < 7; ++kt) {
        FragBF pa;                                   // A layout from LDS, 2 x ds_load_b128
        const __bf16* pr = pbuf + lo * 224 + kt * 32;
        pa.q[0] = ((const uint4*)(pr + 8 * hi))[0];
        pa.q[1] = ((const uint4*)(pr + 16 + 8 * hi))[0];
#pragma unroll
        for (int od = 0; od < 2; ++od) {
            const int col = od * 16 + lo;
            FragBF vf;                               // B layout: K strided in vbuf
#pragma unroll
            for (int i = 0; i < 16; ++i)
                vf.h[i] = vbuf[(kt * 32 + 8 * hi + (i & 7) + 16 * (i >> 3)) * HD + col];
            v8f c;
#pragma unroll
            for (int r = 0; r < 8; ++r) c[r] = oacc[od][r];
            c = wmma_bf16(pa.v, vf.v, c);
#pragma unroll
            for (int r = 0; r < 8; ++r) oacc[od][r] = c[r];
        }
    }

    // store to pre-projection workspace [B, N, C] (C = h*32 + d), bf16
#pragma unroll
    for (int od = 0; od < 2; ++od)
#pragma unroll
        for (int r = 0; r < 8; ++r) {
            const int M = m0 + r + 8 * hi;
            if (M < NTOK)
                ows[((size_t)b * NTOK + M) * CCH + h * HD + od * 16 + lo] =
                    (__bf16)oacc[od][r];
        }
}

// ---------------------------------------------------------------------------
// Kernel 3: out = aws @ proj_w.T + proj_b  (50176 x 256) x (256 x 256), f32 out
// One wave -> 32x64 tile.
// ---------------------------------------------------------------------------
__global__ __launch_bounds__(32) void proj_kernel(
    const __bf16* __restrict__ aws, const __bf16* __restrict__ wb,
    const float* __restrict__ proj_b, float* __restrict__ out)
{
    const int lane = threadIdx.x & 31;
    const int hi = lane >> 4, lo = lane & 15;
    const int m0 = blockIdx.x * 32;
    const int n0 = blockIdx.y * 64;

    float acc[2][4][8];
#pragma unroll
    for (int mi = 0; mi < 2; ++mi)
#pragma unroll
        for (int j = 0; j < 4; ++j)
#pragma unroll
            for (int r = 0; r < 8; ++r) acc[mi][j][r] = 0.f;

#pragma unroll
    for (int kt = 0; kt < 8; ++kt) {
        v16bf a0 = loadA_bf16(aws + (size_t)(m0 + lo) * CCH + kt * 32, hi);
        v16bf a1 = loadA_bf16(aws + (size_t)(m0 + 16 + lo) * CCH + kt * 32, hi);
#pragma unroll
        for (int j = 0; j < 4; ++j) {
            v16bf b = loadB_bf16(wb + (size_t)(n0 + j * 16 + lo) * CCH + kt * 32, hi);
            v8f c0, c1;
#pragma unroll
            for (int r = 0; r < 8; ++r) { c0[r] = acc[0][j][r]; c1[r] = acc[1][j][r]; }
            c0 = wmma_bf16(a0, b, c0);
            c1 = wmma_bf16(a1, b, c1);
#pragma unroll
            for (int r = 0; r < 8; ++r) { acc[0][j][r] = c0[r]; acc[1][j][r] = c1[r]; }
        }
    }

#pragma unroll
    for (int mi = 0; mi < 2; ++mi)
#pragma unroll
        for (int j = 0; j < 4; ++j) {
            const int col = n0 + j * 16 + lo;
            const float pb = proj_b[col];
#pragma unroll
            for (int r = 0; r < 8; ++r)
                out[(size_t)(m0 + mi * 16 + r + 8 * hi) * CCH + col] = acc[mi][j][r] + pb;
        }
}

// ---------------------------------------------------------------------------
extern "C" void kernel_launch(void* const* d_in, const int* in_sizes, int n_in,
                              void* d_out, int out_size, void* d_ws, size_t ws_size,
                              hipStream_t stream) {
    (void)in_sizes; (void)n_in; (void)out_size; (void)ws_size;
    const float* x          = (const float*)d_in[0];
    const float* mask       = (const float*)d_in[1];
    const float* qkv_w      = (const float*)d_in[2];
    const float* qkv_b      = (const float*)d_in[3];
    const float* proj_w     = (const float*)d_in[4];
    const float* proj_b     = (const float*)d_in[5];
    const float* bias_table = (const float*)d_in[6];
    const int*   rel_index  = (const int*)d_in[7];

    // workspace (bf16 halfs):
    //   q,k,v [B,H,N,32] : 3 * 12,845,056
    //   aws   [B,N,C]    : 12,845,056
    //   xb    [B*N, C]   : 12,845,056
    //   wqkv  [768,256]  : 196,608
    //   wproj [256,256]  : 65,536
    const size_t per = (size_t)B_WIN * NHEAD * NTOK * HD;   // 12,845,056 halfs
    __bf16* qws   = (__bf16*)d_ws;
    __bf16* kws   = qws + per;
    __bf16* vws   = kws + per;
    __bf16* aws   = vws + per;
    __bf16* xb    = aws + per;
    __bf16* wqkv  = xb + per;
    __bf16* wproj = wqkv + (size_t)768 * CCH;

    // Stage 0: one-pass f32 -> bf16 conversions (coalesced, done once).
    {
        const int nx = (NROWS * CCH) / 8;                    // 1,605,632
        cvt_bf16_kernel<<<(nx + 255) / 256, 256, 0, stream>>>(x, xb, nx);
        const int nw = (768 * CCH) / 8;                      // 24,576
        cvt_bf16_kernel<<<(nw + 255) / 256, 256, 0, stream>>>(qkv_w, wqkv, nw);
        const int np = (CCH * CCH) / 8;                      // 8,192
        cvt_bf16_kernel<<<(np + 255) / 256, 256, 0, stream>>>(proj_w, wproj, np);
    }

    qkv_kernel <<<dim3(NROWS / 32, 12),   32, 0, stream>>>(xb, wqkv, qkv_b,
                                                           qws, kws, vws);
    attn_kernel<<<dim3(13, NHEAD, B_WIN), 32, 0, stream>>>(qws, kws, vws, mask,
                                                           bias_table, rel_index, aws);
    proj_kernel<<<dim3(NROWS / 32, 4),    32, 0, stream>>>(aws, wproj, proj_b,
                                                           (float*)d_out);
}
